// CANN_75857712382071
// MI455X (gfx1250) — matrix operand order repaired and checked by reading, not compiled, not run on hardware
//
#include <hip/hip_runtime.h>
#include <hip/hip_bf16.h>

// Problem constants (fixed by the reference).
constexpr int BATCH = 32;
constexpr int ALEN  = 2048;
constexpr int DIM   = 128;

typedef __attribute__((ext_vector_type(16))) __bf16 v16bf;
typedef __attribute__((ext_vector_type(8)))  __bf16 v8bf;
typedef __attribute__((ext_vector_type(8)))  float  v8f;

typedef unsigned int u32x4 __attribute__((ext_vector_type(4)));
typedef int          i32x8 __attribute__((ext_vector_type(8)));
typedef int          i32x4 __attribute__((ext_vector_type(4)));

union FragA {
    v16bf v;
    v8bf  h[2];
};

__device__ inline v8bf load_cvt8(const float* __restrict__ p) {
    v8bf r;
#pragma unroll
    for (int i = 0; i < 8; ++i) r[i] = (__bf16)p[i];
    return r;
}

__device__ inline v16bf load_cvt16(const float* __restrict__ p) {
    v16bf r;
#pragma unroll
    for (int i = 0; i < 16; ++i) r[i] = (__bf16)p[i];
    return r;
}

// -----------------------------------------------------------------------------
// Tensor Data Mover: 2D tile (global -> LDS), bf16 elements, row padding so
// LDS rows land on conflict-free strides.  Wave-uniform; EXEC ignored.
// This toolchain exposes the 6-arg builtin:
//   (uint32x4 g0, int32x8 g1, int32x4 g2, int32x4 g3, int32x8, i32 cpol)
// -----------------------------------------------------------------------------
__device__ inline void tdm_load_2d(unsigned lds_off, const void* gptr,
                                   unsigned tensor_d0, unsigned tensor_d1,
                                   unsigned tile_d0, unsigned tile_d1,
                                   unsigned long long stride0_elems,
                                   unsigned pad_interval, unsigned pad_amount)
{
    const unsigned long long ga = (unsigned long long)(uintptr_t)gptr;
    u32x4 g0;
    g0[0] = 1u;                                            // count=1, user mode
    g0[1] = lds_off;                                       // LDS byte address
    g0[2] = (unsigned)(ga & 0xFFFFFFFFu);                  // global_addr[31:0]
    g0[3] = (unsigned)((ga >> 32) & 0x1FFFFFFu) | (2u << 30); // [56:32] | type=2
    i32x8 g1;
    g1[0] = (int)((1u << 16)                               // data_size = 2B
                | (1u << 20)                               // pad_enable
                | (pad_interval << 22) | (pad_amount << 25));
    g1[1] = (int)((tensor_d0 & 0xFFFFu) << 16);            // tensor_dim0 lo
    g1[2] = (int)((tensor_d0 >> 16) | ((tensor_d1 & 0xFFFFu) << 16));
    g1[3] = (int)((tensor_d1 >> 16) | (tile_d0 << 16));    // tile_dim0
    g1[4] = (int)tile_d1;                                  // tile_dim1 (dim2=0)
    g1[5] = (int)(stride0_elems & 0xFFFFFFFFull);          // dim0 stride lo
    g1[6] = (int)((stride0_elems >> 32) & 0xFFFFull);      // stride hi | s1=0
    g1[7] = 0;
    const i32x4 z4 = {0, 0, 0, 0};                         // 2D: groups 2/3 off
    const i32x8 z8 = {0, 0, 0, 0, 0, 0, 0, 0};
    __builtin_amdgcn_tensor_load_to_lds(g0, g1, z4, z4, z8, 0);
}

// -----------------------------------------------------------------------------
// Kernel 1: QKV projection.  q = (z @ Wq^T + bq) * 1/sqrt(D)  (scale folded),
// k = z @ Wk^T + bk, v = z @ Wv^T + bv.  One wave: 16 tokens x 128 outputs.
// Writes Q,K row-major bf16 and V transposed (Vt[d][a]) bf16.
// -----------------------------------------------------------------------------
__global__ __launch_bounds__(256) void cann_qkv_proj(
    const float* __restrict__ z,
    const float* __restrict__ Wq, const float* __restrict__ bq,
    const float* __restrict__ Wk, const float* __restrict__ bk,
    const float* __restrict__ Wv, const float* __restrict__ bv,
    __bf16* __restrict__ Qws, __bf16* __restrict__ Kws,
    __bf16* __restrict__ Vtws)
{
    const int lane = threadIdx.x & 31;
    const int wid  = (blockIdx.x * blockDim.x + threadIdx.x) >> 5;
    const int tilesPerBatch = ALEN / 16;           // 128
    const int b  = wid / tilesPerBatch;
    const int a0 = (wid % tilesPerBatch) * 16;
    const int n  = lane & 15;
    const int hl = lane >> 4;
    const float sscale = 0.08838834764831845f;     // 1/sqrt(128)

    FragA zf[4];
    const float* zrow = z + ((size_t)b * ALEN + (a0 + n)) * DIM;
#pragma unroll
    for (int c = 0; c < 4; ++c) {
        zf[c].h[0] = load_cvt8(zrow + 32 * c + 16 * hl);
        zf[c].h[1] = load_cvt8(zrow + 32 * c + 16 * hl + 8);
    }

    const float* Ws[3] = {Wq, Wk, Wv};
    const float* bs[3] = {bq, bk, bv};
    const size_t base  = (size_t)b * ALEN * DIM;

#pragma unroll
    for (int p = 0; p < 3; ++p) {
        const float osc = (p == 0) ? sscale : 1.0f;
#pragma unroll
        for (int et = 0; et < 8; ++et) {
            const int e = et * 16 + n;
            const float* wrow = Ws[p] + (size_t)e * DIM;
            const float bias  = bs[p][e];
            v8f acc = {};
#pragma unroll
            for (int c = 0; c < 4; ++c) {
                v16bf bf = load_cvt16(wrow + 32 * c + 16 * hl);
                acc = __builtin_amdgcn_wmma_f32_16x16x32_bf16(
                    false, zf[c].v, false, bf, (short)0, acc, false, false);
            }
            if (p < 2) {
                __bf16* dst = (p == 0 ? Qws : Kws);
#pragma unroll
                for (int j = 0; j < 8; ++j) {
                    const int row = a0 + j + 8 * hl;
                    dst[base + (size_t)row * DIM + (et * 16 + n)] =
                        (__bf16)((acc[j] + bias) * osc);
                }
            } else {
                v8bf vv;
#pragma unroll
                for (int j = 0; j < 8; ++j) vv[j] = (__bf16)(acc[j] + bias);
                *(v8bf*)(Vtws + base + (size_t)(et * 16 + n) * ALEN + a0 + 8 * hl) = vv;
            }
        }
    }
}

// -----------------------------------------------------------------------------
// Kernel 2: flash attention + fused output projection.
// Block = 8 waves; each wave owns 32 query rows (2 groups of 16 sharing every
// K/V fragment).  K/Vt tiles (32 tokens) staged into LDS by the TDM, double
// buffered.  Row-sums via WMMA against an all-ones fragment.
// -----------------------------------------------------------------------------
// LDS layout (bytes): K tiles 2 x 32x272, V tiles 2 x 128x80, P 8 x 16x80.
constexpr unsigned KTILE_BYTES = 32u * 272u;    // 8704
constexpr unsigned VTILE_BYTES = 128u * 80u;    // 10240
constexpr unsigned KOFF0 = 0u,      KOFF1 = KTILE_BYTES;            // 0, 8704
constexpr unsigned VOFF0 = 17408u,  VOFF1 = VOFF0 + VTILE_BYTES;    // 17408, 27648
constexpr unsigned POFF  = VOFF1 + VTILE_BYTES;                     // 37888
constexpr unsigned SMEM_BYTES = POFF + 8u * 1280u;                  // 48128

__global__ __launch_bounds__(256) void cann_flash_attn(
    const __bf16* __restrict__ Qws, const __bf16* __restrict__ Kws,
    const __bf16* __restrict__ Vtws,
    const float* __restrict__ Wo, const float* __restrict__ bo,
    float* __restrict__ out)
{
    __shared__ __attribute__((aligned(128))) unsigned char smem[SMEM_BYTES];

    const int lane  = threadIdx.x & 31;
    const int wslot = threadIdx.x >> 5;
    const int wid   = (blockIdx.x * blockDim.x + threadIdx.x) >> 5;
    const int wavesPerBatch = ALEN / 32;           // 64
    const int b  = wid / wavesPerBatch;
    const int a0 = (wid % wavesPerBatch) * 32;     // 32 query rows per wave
    const int n  = lane & 15;
    const int hl = lane >> 4;

    const __bf16* Kbase = Kws  + (size_t)b * ALEN * DIM;
    const __bf16* Vbase = Vtws + (size_t)b * ALEN * DIM;   // Vt: [d][a]

    // Q fragments for both row groups (A-operand, 4 chunks of K=32 over d).
    FragA qf[2][4];
#pragma unroll
    for (int g = 0; g < 2; ++g) {
        const __bf16* qrow = Qws + ((size_t)b * ALEN + a0 + g * 16 + n) * DIM;
#pragma unroll
        for (int c = 0; c < 4; ++c) {
            qf[g][c].h[0] = *(const v8bf*)(qrow + 32 * c + 16 * hl);
            qf[g][c].h[1] = *(const v8bf*)(qrow + 32 * c + 16 * hl + 8);
        }
    }

    const v8f zero = {};
    v8f Oacc[2][8];
#pragma unroll
    for (int g = 0; g < 2; ++g)
#pragma unroll
        for (int dt = 0; dt < 8; ++dt) Oacc[g][dt] = zero;

    float m_[2][8], l_[2][8];
#pragma unroll
    for (int g = 0; g < 2; ++g)
#pragma unroll
        for (int j = 0; j < 8; ++j) { m_[g][j] = -1.0e30f; l_[g][j] = 0.0f; }

    v16bf ones;
#pragma unroll
    for (int i = 0; i < 16; ++i) ones[i] = (__bf16)1.0f;

    const float LOG2E = 1.4426950408889634f;
    const unsigned pbase = POFF + (unsigned)wslot * 1280u;
    unsigned char* const pbuf = smem + pbase;

    // Prologue: stage tile 0 (TDM pad: K rows 256B->+16B, V rows 64B->+16B).
    if (wslot == 0) {
        tdm_load_2d((unsigned)(uintptr_t)(smem + KOFF0), Kbase,
                    128u, 32u, 128u, 32u, 128ull, 5u, 3u);
        tdm_load_2d((unsigned)(uintptr_t)(smem + VOFF0), Vbase,
                    32u, 128u, 32u, 128u, 2048ull, 3u, 3u);
    }

    const int NIT = ALEN / 32;                      // 64 steps of 32 tokens
    for (int it = 0; it < NIT; ++it) {
        const int t   = it * 32;
        const int cur = it & 1;
        if (wslot == 0) {
            if (it + 1 < NIT) {   // prefetch next tile into the other buffer
                tdm_load_2d((unsigned)(uintptr_t)(smem + (cur ? KOFF0 : KOFF1)),
                            Kbase + (size_t)(t + 32) * DIM,
                            128u, 32u, 128u, 32u, 128ull, 5u, 3u);
                tdm_load_2d((unsigned)(uintptr_t)(smem + (cur ? VOFF0 : VOFF1)),
                            Vbase + (t + 32),
                            32u, 128u, 32u, 128u, 2048ull, 3u, 3u);
                __builtin_amdgcn_s_wait_tensorcnt(2);  // current tile complete
            } else {
                __builtin_amdgcn_s_wait_tensorcnt(0);
            }
        }
        __syncthreads();

        const unsigned char* kbuf = smem + (cur ? KOFF1 : KOFF0);
        const unsigned char* vbuf = smem + (cur ? VOFF1 : VOFF0);

        // ---- scores: S[g][s] for tokens t+16s, both row groups share kf ----
        v8f S[2][2];
        S[0][0] = zero; S[0][1] = zero; S[1][0] = zero; S[1][1] = zero;
#pragma unroll
        for (int s = 0; s < 2; ++s) {
#pragma unroll
            for (int c = 0; c < 4; ++c) {
                const unsigned char* ka = kbuf + (s * 16 + n) * 272 + 64 * c + 32 * hl;
                FragA kf;
                kf.h[0] = *(const v8bf*)(ka);
                kf.h[1] = *(const v8bf*)(ka + 16);
                S[0][s] = __builtin_amdgcn_wmma_f32_16x16x32_bf16(
                    false, qf[0][c].v, false, kf.v, (short)0, S[0][s], false, false);
                S[1][s] = __builtin_amdgcn_wmma_f32_16x16x32_bf16(
                    false, qf[1][c].v, false, kf.v, (short)0, S[1][s], false, false);
            }
        }

        // ---- online softmax per group; P via LDS; row-sum via WMMA(ones) ----
        FragA pf[2];
#pragma unroll
        for (int g = 0; g < 2; ++g) {
            float alpha_r[8];
#pragma unroll
            for (int j = 0; j < 8; ++j) {
                const float sa = S[g][0][j];
                const float sb = S[g][1][j];
                float rm = fmaxf(sa, sb);
#pragma unroll
                for (int off = 1; off < 16; off <<= 1)
                    rm = fmaxf(rm, __shfl_xor(rm, off, 16));
                const float mnew = fmaxf(m_[g][j], rm);
                alpha_r[j] = exp2f((m_[g][j] - mnew) * LOG2E);
                m_[g][j]   = mnew;
                const __bf16 p0 = (__bf16)exp2f((sa - mnew) * LOG2E);
                const __bf16 p1 = (__bf16)exp2f((sb - mnew) * LOG2E);
                unsigned char* pr = pbuf + (j + 8 * hl) * 80;
                *(__bf16*)(pr + 2 * n)      = p0;    // col n      (token t+n)
                *(__bf16*)(pr + 32 + 2 * n) = p1;    // col 16+n
#pragma unroll
                for (int dt = 0; dt < 8; ++dt) Oacc[g][dt][j] *= alpha_r[j];
            }
            asm volatile("s_wait_dscnt 0" ::: "memory");  // DS in-order per wave
            const unsigned char* pr = pbuf + n * 80 + 16 * hl;
            pf[g].h[0] = *(const v8bf*)(pr);
            pf[g].h[1] = *(const v8bf*)(pr + 32);
            v8f rs = __builtin_amdgcn_wmma_f32_16x16x32_bf16(
                false, pf[g].v, false, ones, (short)0, zero, false, false);
#pragma unroll
            for (int j = 0; j < 8; ++j)
                l_[g][j] = l_[g][j] * alpha_r[j] + rs[j];
        }

        // ---- O[g] += P[g](16x32) @ V(32x128); vf shared by both groups ----
#pragma unroll
        for (int dt = 0; dt < 8; ++dt) {
            const unsigned char* va = vbuf + (dt * 16 + n) * 80 + 16 * hl;
            FragA vf;
            vf.h[0] = *(const v8bf*)(va);
            vf.h[1] = *(const v8bf*)(va + 32);
            Oacc[0][dt] = __builtin_amdgcn_wmma_f32_16x16x32_bf16(
                false, pf[0].v, false, vf.v, (short)0, Oacc[0][dt], false, false);
            Oacc[1][dt] = __builtin_amdgcn_wmma_f32_16x16x32_bf16(
                false, pf[1].v, false, vf.v, (short)0, Oacc[1][dt], false, false);
        }
        __syncthreads();   // everyone done with this tile before it is reloaded
    }

    // ---- epilogue: out = (O / l) . Wo + bo ----
    float wo[8];
#pragma unroll
    for (int dt = 0; dt < 8; ++dt) wo[dt] = Wo[dt * 16 + n];
    const float bov = bo[0];
#pragma unroll
    for (int g = 0; g < 2; ++g) {
#pragma unroll
        for (int j = 0; j < 8; ++j) {
            float partial = 0.0f;
#pragma unroll
            for (int dt = 0; dt < 8; ++dt) partial += Oacc[g][dt][j] * wo[dt];
#pragma unroll
            for (int off = 1; off < 16; off <<= 1)
                partial += __shfl_xor(partial, off, 16);
            if (n == 0) {
                const int row = a0 + g * 16 + j + 8 * hl;
                out[(size_t)b * ALEN + row] = partial / l_[g][j] + bov;
            }
        }
    }
}

// -----------------------------------------------------------------------------
extern "C" void kernel_launch(void* const* d_in, const int* in_sizes, int n_in,
                              void* d_out, int out_size, void* d_ws, size_t ws_size,
                              hipStream_t stream) {
    const float* z  = (const float*)d_in[0];
    const float* Wq = (const float*)d_in[1];
    const float* bq = (const float*)d_in[2];
    const float* Wk = (const float*)d_in[3];
    const float* bk = (const float*)d_in[4];
    const float* Wv = (const float*)d_in[5];
    const float* bv = (const float*)d_in[6];
    const float* Wo = (const float*)d_in[7];
    const float* bo = (const float*)d_in[8];
    float* out = (float*)d_out;

    const size_t elems = (size_t)BATCH * ALEN * DIM;   // 8,388,608
    __bf16* Qws = (__bf16*)d_ws;
    __bf16* Kws = Qws + elems;
    __bf16* Vt  = Kws + elems;                         // total 50.3 MB bf16

    const dim3 blk(256);                               // 8 waves/block
    cann_qkv_proj<<<dim3(BATCH * (ALEN / 16) / 8), blk, 0, stream>>>(
        z, Wq, bq, Wk, bk, Wv, bv, Qws, Kws, Vt);
    cann_flash_attn<<<dim3(BATCH * (ALEN / 32) / 8), blk, 0, stream>>>(
        Qws, Kws, Vt, Wo, bo, out);
}
